// DeepSeekMLA_85624468013208
// MI455X (gfx1250) — matrix operand order repaired and checked by reading, not compiled, not run on hardware
//
#include <hip/hip_runtime.h>

// ---------------- problem constants ----------------
#define T_SEQ   2048
#define DMODEL  2048
#define NHEAD   16
#define NOPE    128
#define ROPE_D  64
#define VDIM    128
#define QKD     192
#define Q_LORA  768
#define KV_LORA 512
#define EPSV    1e-6f
#define SOFTSCALE (-96.0f)   /* reference: QKD * -0.5 */

typedef unsigned short u16;
typedef __attribute__((ext_vector_type(16))) __bf16 v16bf;
typedef __attribute__((ext_vector_type(8)))  __bf16 v8bf;
typedef __attribute__((ext_vector_type(8)))  float  v8f;
typedef __attribute__((ext_vector_type(4)))  unsigned int u32x4;
typedef __attribute__((ext_vector_type(8)))  int          i32x8;
typedef __attribute__((ext_vector_type(4)))  int          i32x4;

__device__ __forceinline__ u16 f2bf(float f) {
  unsigned int u = __float_as_uint(f);
  u += 0x7fffu + ((u >> 16) & 1u);      // round-to-nearest-even
  return (u16)(u >> 16);
}

__device__ __forceinline__ unsigned ldsOff(const void* p) {
  return (unsigned)(unsigned long long)p;   // low 32 bits of generic ptr = LDS byte offset
}

// ---- Tensor Data Mover: 2D tile load (global -> LDS), D# per ISA 8.3/8.4 ----
// dims relative to the tile base so hardware zero-fills OOB rows/cols.
// 6-arg builtin form: (g0, g1, g2, g3, g_extra, cpol)
__device__ __forceinline__ void tdm_load_2d(unsigned lds_off, const void* gaddr,
                                            unsigned td0,        // tensor_dim0 bound (elems)
                                            unsigned td1,        // tensor_dim1 bound (rows)
                                            unsigned tile_w,     // tile_dim0 (elems)
                                            unsigned tile_h,     // tile_dim1 (rows)
                                            unsigned stride) {   // tensor_dim0_stride (elems)
  unsigned long long ga = (unsigned long long)gaddr;
  u32x4 g0;
  g0[0] = 1u;                                   // count=1, user mode
  g0[1] = lds_off;                              // lds_addr (bytes)
  g0[2] = (unsigned)ga;                         // global_addr[31:0]
  g0[3] = (unsigned)((ga >> 32) & 0x01FFFFFFu) | (2u << 30);  // addr[56:32] | type=2
  i32x8 g1;
  g1[0] = (int)(1u << 16);                      // data_size=1 (2 bytes)
  g1[1] = (int)((td0 & 0xFFFFu) << 16);         // tensor_dim0[15:0]
  g1[2] = (int)((td0 >> 16) | ((td1 & 0xFFFFu) << 16));
  g1[3] = (int)((td1 >> 16) | (tile_w << 16));  // tile_dim0
  g1[4] = (int)(tile_h & 0xFFFFu);              // tile_dim1 (tile_dim2=0)
  g1[5] = (int)stride;                          // tensor_dim0_stride[31:0]
  g1[6] = 0;                                    // stride[47:32]=0, dim1_stride=0
  g1[7] = 0;
  i32x4 z4; z4[0] = z4[1] = z4[2] = z4[3] = 0;
  i32x8 z8; z8[0]=z8[1]=z8[2]=z8[3]=z8[4]=z8[5]=z8[6]=z8[7]=0;
  __builtin_amdgcn_tensor_load_to_lds(g0, g1, z4, z4, z8, 0);
}

// ---- WMMA fragment helpers (layouts per CDNA5 ISA 7.12.2, wave32) ----
__device__ __forceinline__ v16bf load_frag_a(const u16* base, int ld, int lane, int koff) {
  const int m = lane & 15, hf = lane >> 4;
  const u16* p = base + m * ld + koff + hf * 8;
  v8bf lo = *reinterpret_cast<const v8bf*>(p);
  v8bf hi = *reinterpret_cast<const v8bf*>(p + 16);
  return __builtin_shufflevector(lo, hi, 0,1,2,3,4,5,6,7,8,9,10,11,12,13,14,15);
}
__device__ __forceinline__ v16bf load_frag_b(const u16* base, int ld, int lane, int koff) {
  const int n = lane & 15, hf = lane >> 4;
  const u16* p = base + n * ld + koff + hf * 8;
  v8bf lo = *reinterpret_cast<const v8bf*>(p);
  v8bf hi = *reinterpret_cast<const v8bf*>(p + 16);
  return __builtin_shufflevector(lo, hi, 0,1,2,3,4,5,6,7,8,9,10,11,12,13,14,15);
}
__device__ __forceinline__ v8f wmma_bf16(v16bf a, v16bf b, v8f c) {
  return __builtin_amdgcn_wmma_f32_16x16x32_bf16(false, a, false, b, (short)0, c, false, false);
}

// ---------------- fp32 -> bf16 convert ----------------
__global__ void k_cvt_bf16(const float* __restrict__ in, u16* __restrict__ out, int n) {
  int i = blockIdx.x * blockDim.x + threadIdx.x;
  if (i < n) out[i] = f2bf(in[i]);
}

// ---------------- generic GEMM: C[f32, MxN] = A[bf16, MxK] * W[bf16, NxK]^T ----------------
// 256 thr = 8 waves (2x4); block tile 64x128, KT=64; TDM-staged, double-buffered LDS.
__global__ void k_gemm(const u16* __restrict__ A, const u16* __restrict__ W,
                       float* __restrict__ C, int M, int N, int K) {
  __shared__ __align__(16) u16 sA[2][64 * 64];    // 2 x 8KB
  __shared__ __align__(16) u16 sB[2][128 * 64];   // 2 x 16KB
  const int tid = threadIdx.x;
  const int lane = tid & 31, wave = tid >> 5;
  const int wr = wave >> 2, wc = wave & 3;        // 2 x 4 wave grid
  const int m0 = blockIdx.y * 64, n0 = blockIdx.x * 128;
  const int nIn = lane & 15, hf = lane >> 4;

  auto stage = [&](int k0, int buf) {
    if (wave == 0)
      tdm_load_2d(ldsOff(&sA[buf][0]), &A[(size_t)m0 * K + k0],
                  (unsigned)(K - k0), (unsigned)(M - m0), 64u, 64u, (unsigned)K);
    else if (wave == 1)
      tdm_load_2d(ldsOff(&sB[buf][0]), &W[(size_t)n0 * K + k0],
                  (unsigned)(K - k0), (unsigned)(N - n0), 64u, 128u, (unsigned)K);
  };

  v8f z8 = {0.f,0.f,0.f,0.f,0.f,0.f,0.f,0.f};
  v8f a00 = z8, a01 = z8, a10 = z8, a11 = z8;

  const int nk = K / 64;
  stage(0, 0);
  for (int t = 0; t < nk; ++t) {
    const int buf = t & 1;
    if (wave < 2) __builtin_amdgcn_s_wait_tensorcnt(0);
    __syncthreads();                       // tile t resident; everyone done with buf^1
    if (t + 1 < nk) stage((t + 1) * 64, buf ^ 1);   // overlap next DMA with compute
    #pragma unroll
    for (int kk = 0; kk < 64; kk += 32) {
      v16bf fa0 = load_frag_a(&sA[buf][(wr * 32 +  0) * 64], 64, lane, kk);
      v16bf fa1 = load_frag_a(&sA[buf][(wr * 32 + 16) * 64], 64, lane, kk);
      v16bf fb0 = load_frag_b(&sB[buf][(wc * 32 +  0) * 64], 64, lane, kk);
      v16bf fb1 = load_frag_b(&sB[buf][(wc * 32 + 16) * 64], 64, lane, kk);
      a00 = wmma_bf16(fa0, fb0, a00);
      a01 = wmma_bf16(fa0, fb1, a01);
      a10 = wmma_bf16(fa1, fb0, a10);
      a11 = wmma_bf16(fa1, fb1, a11);
    }
  }
  // C layout: VGPR r -> row (r + 8*hf), col nIn
  v8f accs[4] = { a00, a01, a10, a11 };
  #pragma unroll
  for (int i = 0; i < 2; ++i)
    #pragma unroll
    for (int j = 0; j < 2; ++j) {
      v8f acc = accs[i * 2 + j];
      int gm = m0 + wr * 32 + i * 16 + 8 * hf;
      int gn = n0 + wc * 32 + j * 16 + nIn;
      if (gn < N) {
        #pragma unroll
        for (int r = 0; r < 8; ++r)
          C[(size_t)(gm + r) * N + gn] = acc[r];
      }
    }
}

// ---------------- row-wise RMS norm (f32 in -> bf16 out) ----------------
__global__ void k_rmsnorm(const float* __restrict__ in, const float* __restrict__ w,
                          u16* __restrict__ out, int D) {
  const int row = blockIdx.x, tid = threadIdx.x;
  const float* p = in + (size_t)row * D;
  __shared__ float red[256];
  float ss = 0.f;
  for (int i = tid; i < D; i += 256) { float v = p[i]; ss += v * v; }
  red[tid] = ss; __syncthreads();
  for (int s = 128; s > 0; s >>= 1) { if (tid < s) red[tid] += red[tid + s]; __syncthreads(); }
  float scale = rsqrtf(red[0] / (float)D + EPSV);
  for (int i = tid; i < D; i += 256) out[(size_t)row * D + i] = f2bf(p[i] * scale * w[i]);
}

// ---------------- q assemble: cast nope + RoPE on pe -> qh[T][16][192] bf16 ----------------
__global__ void k_q_assemble(const float* __restrict__ qraw,    // T x 3072
                             const float* __restrict__ freqs,   // T x 32
                             u16* __restrict__ qh) {
  const int t = blockIdx.x, tid = threadIdx.x;
  const float* qr = qraw + (size_t)t * (NHEAD * QKD);
  u16* o = qh + (size_t)t * (NHEAD * QKD);
  for (int i = tid; i < NHEAD * NOPE; i += blockDim.x) {
    int h = i >> 7, d = i & 127;
    o[h * QKD + d] = f2bf(qr[h * QKD + d]);
  }
  for (int i = tid; i < NHEAD * (ROPE_D / 2); i += blockDim.x) {
    int h = i >> 5, pr = i & 31;
    float c = cosf(freqs[t * 32 + pr]), s = sinf(freqs[t * 32 + pr]);
    float x1 = qr[h * QKD + NOPE + 2 * pr], x2 = qr[h * QKD + NOPE + 2 * pr + 1];
    o[h * QKD + NOPE + 2 * pr]     = f2bf(x1 * c - x2 * s);
    o[h * QKD + NOPE + 2 * pr + 1] = f2bf(x1 * s + x2 * c);
  }
}

// ---------------- kv split: RMS(first 512) -> bf16; RoPE(last 64) -> k_pe bf16 ----------------
__global__ void k_kv_split(const float* __restrict__ kvf,     // T x 576
                           const float* __restrict__ w,       // 512
                           const float* __restrict__ freqs,   // T x 32
                           u16* __restrict__ ckv,             // T x 512
                           u16* __restrict__ kpe) {           // T x 64
  const int t = blockIdx.x, tid = threadIdx.x;
  const float* p = kvf + (size_t)t * (KV_LORA + ROPE_D);
  __shared__ float red[256];
  float ss = 0.f;
  for (int i = tid; i < KV_LORA; i += 256) { float v = p[i]; ss += v * v; }
  red[tid] = ss; __syncthreads();
  for (int s = 128; s > 0; s >>= 1) { if (tid < s) red[tid] += red[tid + s]; __syncthreads(); }
  float scale = rsqrtf(red[0] / (float)KV_LORA + EPSV);
  for (int i = tid; i < KV_LORA; i += 256) ckv[(size_t)t * KV_LORA + i] = f2bf(p[i] * scale * w[i]);
  if (tid < ROPE_D / 2) {
    float c = cosf(freqs[t * 32 + tid]), s = sinf(freqs[t * 32 + tid]);
    float x1 = p[KV_LORA + 2 * tid], x2 = p[KV_LORA + 2 * tid + 1];
    kpe[(size_t)t * ROPE_D + 2 * tid]     = f2bf(x1 * c - x2 * s);
    kpe[(size_t)t * ROPE_D + 2 * tid + 1] = f2bf(x1 * s + x2 * c);
  }
}

// ---------------- kv assemble: kh[T][16][192]; V pre-transposed vT[16][128][T] ----------------
__global__ void k_kv_assemble(const float* __restrict__ kvraw, // T x 16 x 256
                              const u16* __restrict__ kpe,     // T x 64
                              u16* __restrict__ kh,
                              u16* __restrict__ vT) {          // [h][d][t]
  const int t = blockIdx.x, tid = threadIdx.x;
  const float* p = kvraw + (size_t)t * (NHEAD * (NOPE + VDIM));
  u16* ko = kh + (size_t)t * (NHEAD * QKD);
  for (int i = tid; i < NHEAD * NOPE; i += blockDim.x) {
    int h = i >> 7, d = i & 127;
    ko[h * QKD + d] = f2bf(p[h * (NOPE + VDIM) + d]);
    vT[((size_t)h * VDIM + d) * T_SEQ + t] = f2bf(p[h * (NOPE + VDIM) + NOPE + d]);
  }
  for (int i = tid; i < NHEAD * ROPE_D; i += blockDim.x) {
    int h = i >> 6, d = i & 63;
    ko[h * QKD + NOPE + d] = kpe[(size_t)t * ROPE_D + d];
  }
}

// ---------------- flash attention: block = (64 queries, 1 head), 4 waves ----------------
__global__ void k_flash(const u16* __restrict__ qh, const u16* __restrict__ kh,
                        const u16* __restrict__ vT, u16* __restrict__ y) {
  const int h  = blockIdx.x;        // 16 heads
  const int qb = blockIdx.y;        // T/64 query blocks
  const int tid = threadIdx.x;      // 128
  const int wave = tid >> 5, lane = tid & 31;
  const int nIn = lane & 15, hf = lane >> 4;

  __shared__ __align__(16) u16 sQ[64 * QKD];        // 24KB
  __shared__ __align__(16) u16 sK[2][32 * QKD];     // 2 x 12KB
  __shared__ __align__(16) u16 sVt[2][VDIM * 32];   // 2 x  8KB ([d][key])
  __shared__ __align__(16) u16 sP[4][16 * 32];      //  4KB (per-wave P tile)

  auto stage_kv = [&](int kt, int buf) {
    const int kb = kt * 32;
    if (wave == 0)
      tdm_load_2d(ldsOff(&sK[buf][0]), &kh[((size_t)kb * NHEAD + h) * QKD],
                  (unsigned)QKD, (unsigned)(T_SEQ - kb), (unsigned)QKD, 32u,
                  (unsigned)(NHEAD * QKD));
    else if (wave == 1)
      tdm_load_2d(ldsOff(&sVt[buf][0]), &vT[((size_t)h * VDIM) * T_SEQ + kb],
                  (unsigned)(T_SEQ - kb), (unsigned)VDIM, 32u, (unsigned)VDIM,
                  (unsigned)T_SEQ);
  };

  // prologue: DMA Q block + first K/V tile
  if (wave == 2)
    tdm_load_2d(ldsOff(sQ), &qh[(((size_t)qb * 64) * NHEAD + h) * QKD],
                (unsigned)QKD, (unsigned)(T_SEQ - qb * 64), (unsigned)QKD, 64u,
                (unsigned)(NHEAD * QKD));
  stage_kv(0, 0);
  if (wave < 3) __builtin_amdgcn_s_wait_tensorcnt(0);
  __syncthreads();

  // per-wave Q fragments (16 rows x 192)
  v16bf qf[6];
  #pragma unroll
  for (int kk = 0; kk < 6; ++kk)
    qf[kk] = load_frag_a(&sQ[(wave * 16) * QKD], QKD, lane, kk * 32);

  v8f z8 = {0.f,0.f,0.f,0.f,0.f,0.f,0.f,0.f};
  v8f o[8];
  float m_i[8], l_i[8];
  #pragma unroll
  for (int c = 0; c < 8; ++c) o[c] = z8;
  #pragma unroll
  for (int r = 0; r < 8; ++r) { m_i[r] = -1e30f; l_i[r] = 0.f; }

  const int ktEnd = 2 * (qb + 1);           // causal: tiles of 32 keys
  for (int kt = 0; kt < ktEnd; ++kt) {
    const int buf = kt & 1;
    const int kb = kt * 32;
    if (kt + 1 < ktEnd) stage_kv(kt + 1, buf ^ 1);   // overlap next DMA

    // scores: 2 key-column fragments x 6 K-steps
    v8f s0 = z8, s1 = z8;
    #pragma unroll
    for (int kk = 0; kk < 6; ++kk) {
      v16bf b0 = load_frag_b(&sK[buf][0],         QKD, lane, kk * 32);
      v16bf b1 = load_frag_b(&sK[buf][16 * QKD],  QKD, lane, kk * 32);
      s0 = wmma_bf16(qf[kk], b0, s0);
      s1 = wmma_bf16(qf[kk], b1, s1);
    }

    // online softmax (row r of this lane = qb*64 + wave*16 + r + 8*hf)
    float alpha[8];
    #pragma unroll
    for (int r = 0; r < 8; ++r) {
      int q = qb * 64 + wave * 16 + r + 8 * hf;
      float v0 = s0[r] * SOFTSCALE; if (kb + nIn > q)      v0 = -1e30f;
      float v1 = s1[r] * SOFTSCALE; if (kb + 16 + nIn > q) v1 = -1e30f;
      float mx = fmaxf(v0, v1);
      mx = fmaxf(mx, __shfl_xor(mx, 1, 32));
      mx = fmaxf(mx, __shfl_xor(mx, 2, 32));
      mx = fmaxf(mx, __shfl_xor(mx, 4, 32));
      mx = fmaxf(mx, __shfl_xor(mx, 8, 32));
      float mnew = fmaxf(m_i[r], mx);
      float p0 = __expf(v0 - mnew), p1 = __expf(v1 - mnew);
      float ls = p0 + p1;
      ls += __shfl_xor(ls, 1, 32);
      ls += __shfl_xor(ls, 2, 32);
      ls += __shfl_xor(ls, 4, 32);
      ls += __shfl_xor(ls, 8, 32);
      alpha[r] = __expf(m_i[r] - mnew);
      l_i[r] = l_i[r] * alpha[r] + ls;
      m_i[r] = mnew;
      sP[wave][(r + 8 * hf) * 32 + nIn]      = f2bf(p0);
      sP[wave][(r + 8 * hf) * 32 + 16 + nIn] = f2bf(p1);
    }
    #pragma unroll
    for (int c = 0; c < 8; ++c)
      #pragma unroll
      for (int r = 0; r < 8; ++r) o[c][r] *= alpha[r];

    // PV: P(16x32) x V(32x128); sP is intra-wave (LDS ops in-order per wave)
    v16bf pf = load_frag_a(sP[wave], 32, lane, 0);
    #pragma unroll
    for (int c = 0; c < 8; ++c) {
      v16bf bv = load_frag_b(&sVt[buf][(c * 16) * 32], 32, lane, 0);
      o[c] = wmma_bf16(pf, bv, o[c]);
    }

    if (wave < 2) __builtin_amdgcn_s_wait_tensorcnt(0);
    __syncthreads();       // next tile resident; all waves done with buf
  }

  // epilogue: normalize + store y[T][16][128] bf16
  #pragma unroll
  for (int c = 0; c < 8; ++c) {
    #pragma unroll
    for (int r = 0; r < 8; ++r) {
      int q = qb * 64 + wave * 16 + r + 8 * hf;
      float val = o[c][r] / l_i[r];
      y[((size_t)q * NHEAD + h) * VDIM + c * 16 + nIn] = f2bf(val);
    }
  }
}

// ---------------- host-side launch ----------------
extern "C" void kernel_launch(void* const* d_in, const int* in_sizes, int n_in,
                              void* d_out, int out_size, void* d_ws, size_t ws_size,
                              hipStream_t stream) {
  (void)in_sizes; (void)n_in; (void)out_size; (void)ws_size;
  const float* x        = (const float*)d_in[0];
  const float* freqs    = (const float*)d_in[1];
  /* d_in[2] = mask : causality applied analytically */
  const float* wq_a     = (const float*)d_in[3];
  const float* q_norm_w = (const float*)d_in[4];
  const float* wq_b     = (const float*)d_in[5];
  const float* wkv_a    = (const float*)d_in[6];
  const float* kv_norm_w= (const float*)d_in[7];
  const float* wkv_b    = (const float*)d_in[8];
  const float* wo       = (const float*)d_in[9];
  float* out = (float*)d_out;

  // carve workspace
  char* wp = (char*)d_ws;
  auto carve = [&](size_t bytes) { char* p = wp; wp += (bytes + 255) & ~(size_t)255; return p; };
  u16* xb    = (u16*)carve((size_t)T_SEQ * DMODEL * 2);
  u16* wqab  = (u16*)carve((size_t)Q_LORA * DMODEL * 2);
  u16* wqbb  = (u16*)carve((size_t)NHEAD * QKD * Q_LORA * 2);
  u16* wkvab = (u16*)carve((size_t)(KV_LORA + ROPE_D) * DMODEL * 2);
  u16* wkvbb = (u16*)carve((size_t)NHEAD * (NOPE + VDIM) * KV_LORA * 2);
  u16* wob   = (u16*)carve((size_t)DMODEL * NHEAD * VDIM * 2);
  float* scratch = (float*)carve((size_t)T_SEQ * 4096 * 4);   // reused f32 GEMM output
  u16* qcn = (u16*)carve((size_t)T_SEQ * Q_LORA * 2);
  u16* qhb = (u16*)carve((size_t)T_SEQ * NHEAD * QKD * 2);
  u16* ckv = (u16*)carve((size_t)T_SEQ * KV_LORA * 2);
  u16* kpe = (u16*)carve((size_t)T_SEQ * ROPE_D * 2);
  u16* khb = (u16*)carve((size_t)T_SEQ * NHEAD * QKD * 2);
  u16* vTb = (u16*)carve((size_t)NHEAD * VDIM * T_SEQ * 2);
  u16* yb  = (u16*)carve((size_t)T_SEQ * NHEAD * VDIM * 2);

  auto cvt = [&](const float* src, u16* dst, int n) {
    k_cvt_bf16<<<(n + 255) / 256, 256, 0, stream>>>(src, dst, n);
  };
  auto gemm = [&](const u16* A, const u16* W, float* C, int M, int N, int K) {
    dim3 grid((N + 127) / 128, M / 64);
    k_gemm<<<grid, 256, 0, stream>>>(A, W, C, M, N, K);
  };

  // 1) convert operands to bf16
  cvt(x,     xb,    T_SEQ * DMODEL);
  cvt(wq_a,  wqab,  Q_LORA * DMODEL);
  cvt(wq_b,  wqbb,  NHEAD * QKD * Q_LORA);
  cvt(wkv_a, wkvab, (KV_LORA + ROPE_D) * DMODEL);
  cvt(wkv_b, wkvbb, NHEAD * (NOPE + VDIM) * KV_LORA);
  cvt(wo,    wob,   DMODEL * NHEAD * VDIM);

  // 2) q low-rank projection + RMS norm
  gemm(xb, wqab, scratch, T_SEQ, Q_LORA, DMODEL);
  k_rmsnorm<<<T_SEQ, 256, 0, stream>>>(scratch, q_norm_w, qcn, Q_LORA);

  // 3) q up-projection + RoPE
  gemm(qcn, wqbb, scratch, T_SEQ, NHEAD * QKD, Q_LORA);
  k_q_assemble<<<T_SEQ, 256, 0, stream>>>(scratch, freqs, qhb);

  // 4) kv low-rank projection, split: RMS(c_kv) + RoPE(k_pe)
  gemm(xb, wkvab, scratch, T_SEQ, KV_LORA + ROPE_D, DMODEL);
  k_kv_split<<<T_SEQ, 256, 0, stream>>>(scratch, kv_norm_w, freqs, ckv, kpe);

  // 5) kv up-projection + assemble K (nope|pe) and V^T
  gemm(ckv, wkvbb, scratch, T_SEQ, NHEAD * (NOPE + VDIM), KV_LORA);
  k_kv_assemble<<<T_SEQ, 256, 0, stream>>>(scratch, kpe, khb, vTb);

  // 6) flash attention
  {
    dim3 grid(NHEAD, T_SEQ / 64);
    k_flash<<<grid, 128, 0, stream>>>(qhb, khb, vTb, yb);
  }

  // 7) output projection (fp32 result)
  gemm(yb, wob, out, T_SEQ, DMODEL, NHEAD * VDIM);
}
